// Fast_iTPN_28235115004014
// MI455X (gfx1250) — compile-verified, matrix operand order checked by reading, not run on hardware
//
#include <hip/hip_runtime.h>

typedef __attribute__((ext_vector_type(16))) _Float16 v16h;
typedef __attribute__((ext_vector_type(8)))  _Float16 v8h;
typedef __attribute__((ext_vector_type(8)))  float    v8f;

#define MAKE_V16(a, b) __builtin_shufflevector((a), (b), 0,1,2,3,4,5,6,7,8,9,10,11,12,13,14,15)

#define MTOT 12608              // 64 * 197 valid rows
#define MPAD 12672              // 99 * 128
#define KDIM 768
#define NQKV 2304
#define NH   12
#define HD   64
#define NT   197
#define NPAD 224                // 7 * 32
#define LDP  40                 // LDS pitch (f16) for 32-wide K tiles (80B: 16B-aligned rows, conflict-friendly)
#define KSP  72                 // LDS pitch for K tile rows (64 + 8)

// ---------------------------------------------------------------------------
// small utility kernels
// ---------------------------------------------------------------------------
__global__ void zero_f4_kernel(float4* p, int n4) {
    int i = blockIdx.x * blockDim.x + threadIdx.x;
    if (i < n4) p[i] = make_float4(0.f, 0.f, 0.f, 0.f);
}

__global__ void cvt_f16_kernel(const float* __restrict__ s, _Float16* __restrict__ d, int n) {
    int i = blockIdx.x * blockDim.x + threadIdx.x;
    if (i < n) d[i] = (_Float16)s[i];
}

__global__ void cvt_x_pad_kernel(const float* __restrict__ s, _Float16* __restrict__ d,
                                 int rows, int cols, int rowsPad) {
    int i = blockIdx.x * blockDim.x + threadIdx.x;
    int tot = rowsPad * cols;
    if (i >= tot) return;
    int r = i / cols;
    d[i] = (r < rows) ? (_Float16)s[i] : (_Float16)0.f;
}

__global__ void build_rpb_kernel(const float* __restrict__ rh, const float* __restrict__ rw,
                                 const int* __restrict__ hi, const int* __restrict__ wi,
                                 float* __restrict__ dst) {
    int idx = blockIdx.x * blockDim.x + threadIdx.x;
    if (idx >= NH * NPAD * NPAD) return;
    int h = idx / (NPAD * NPAD);
    int r = idx % (NPAD * NPAD);
    int i = r / NPAD, j = r % NPAD;
    float v;
    if (i < NT && j < NT) {
        int a = hi[i * NT + j];
        int b = wi[i * NT + j];
        v = rh[a * NH + h] + rw[b * NH + h];
    } else {
        v = -1.0e30f;   // mask padded key columns
    }
    dst[idx] = v;
}

// ---------------------------------------------------------------------------
// shared WMMA GEMM mainloop: C[128x128] += A[128xK] * W[128xK]^T (W is [n][k])
// 256 threads = 8 waves, wave grid 2(M) x 4(N), wave tile 64x32
// ---------------------------------------------------------------------------
__device__ __forceinline__ void wmma_gemm_mainloop(
    const _Float16* __restrict__ A, const _Float16* __restrict__ W,
    int bm, int bn, _Float16* sa, _Float16* sb, v8f (&acc)[4][2])
{
    const int t = threadIdx.x, lane = t & 31, wid = t >> 5;
    const int wm = (wid >> 2) * 64, wn = (wid & 3) * 32;
    const int half = lane >> 4, lr = lane & 15;
    for (int k0 = 0; k0 < KDIM; k0 += 32) {
        #pragma unroll
        for (int i = 0; i < 2; ++i) {
            int c = t + i * 256;            // 512 16B-chunks per 128x32 tile
            int r = c >> 2, cc = (c & 3) * 8;
            *(v8h*)(sa + r * LDP + cc) = *(const v8h*)(A + (size_t)(bm + r) * KDIM + k0 + cc);
            *(v8h*)(sb + r * LDP + cc) = *(const v8h*)(W + (size_t)(bn + r) * KDIM + k0 + cc);
        }
        __syncthreads();
        v16h af[4], bf[2];
        #pragma unroll
        for (int i = 0; i < 4; ++i) {
            const _Float16* p = sa + (wm + i * 16 + lr) * LDP + 8 * half;
            v8h c1 = *(const v8h*)p, c2 = *(const v8h*)(p + 16);
            af[i] = MAKE_V16(c1, c2);
        }
        #pragma unroll
        for (int j = 0; j < 2; ++j) {
            const _Float16* p = sb + (wn + j * 16 + lr) * LDP + 8 * half;
            v8h c1 = *(const v8h*)p, c2 = *(const v8h*)(p + 16);
            bf[j] = MAKE_V16(c1, c2);
        }
        #pragma unroll
        for (int i = 0; i < 4; ++i)
            #pragma unroll
            for (int j = 0; j < 2; ++j)
                acc[i][j] = __builtin_amdgcn_wmma_f32_16x16x32_f16(
                    false, af[i], false, bf[j], (short)0, acc[i][j], false, false);
        __syncthreads();
    }
}

// ---------------------------------------------------------------------------
// GEMM 1: qkv = x @ qkv_w^T + bias, scatter into q/k/vT head layouts (f16)
// ---------------------------------------------------------------------------
__global__ __launch_bounds__(256) void gemm_qkv_kernel(
    const _Float16* __restrict__ A, const _Float16* __restrict__ W,
    const float* __restrict__ qb, const float* __restrict__ vb,
    _Float16* __restrict__ qo, _Float16* __restrict__ ko, _Float16* __restrict__ vto)
{
    __shared__ _Float16 sa[128 * LDP];
    __shared__ _Float16 sb[128 * LDP];
    const int bm = blockIdx.x * 128, bn = blockIdx.y * 128;
    const int t = threadIdx.x, lane = t & 31, wid = t >> 5;
    const int wm = (wid >> 2) * 64, wn = (wid & 3) * 32;
    const int half = lane >> 4, lr = lane & 15;
    v8f acc[4][2] = {};
    wmma_gemm_mainloop(A, W, bm, bn, sa, sb, acc);
    #pragma unroll
    for (int i = 0; i < 4; ++i) {
        #pragma unroll
        for (int j = 0; j < 2; ++j) {
            #pragma unroll
            for (int v = 0; v < 8; ++v) {
                int m = bm + wm + i * 16 + v + 8 * half;
                if (m >= MTOT) continue;
                int n = bn + wn + j * 16 + lr;
                float x = acc[i][j][v];
                int which = n / 768;
                int r = n - which * 768;
                int h = r >> 6, d = r & 63;
                int b = m / 197, nn = m - b * 197;
                size_t base = (size_t)(b * NH + h);
                if (which == 0) {
                    qo[(base * NPAD + nn) * HD + d] = (_Float16)((x + qb[r]) * 0.125f);
                } else if (which == 1) {
                    ko[(base * NPAD + nn) * HD + d] = (_Float16)x;
                } else {
                    vto[(base * HD + d) * NPAD + nn] = (_Float16)(x + vb[r]);
                }
            }
        }
    }
}

// ---------------------------------------------------------------------------
// GEMM 3: out = attn @ proj_w^T + proj_b (f32 out)
// ---------------------------------------------------------------------------
__global__ __launch_bounds__(256) void gemm_proj_kernel(
    const _Float16* __restrict__ A, const _Float16* __restrict__ W,
    const float* __restrict__ bias, float* __restrict__ out)
{
    __shared__ _Float16 sa[128 * LDP];
    __shared__ _Float16 sb[128 * LDP];
    const int bm = blockIdx.x * 128, bn = blockIdx.y * 128;
    const int t = threadIdx.x, lane = t & 31, wid = t >> 5;
    const int wm = (wid >> 2) * 64, wn = (wid & 3) * 32;
    const int half = lane >> 4, lr = lane & 15;
    v8f acc[4][2] = {};
    wmma_gemm_mainloop(A, W, bm, bn, sa, sb, acc);
    #pragma unroll
    for (int i = 0; i < 4; ++i) {
        #pragma unroll
        for (int j = 0; j < 2; ++j) {
            #pragma unroll
            for (int v = 0; v < 8; ++v) {
                int m = bm + wm + i * 16 + v + 8 * half;
                if (m >= MTOT) continue;
                int n = bn + wn + j * 16 + lr;
                out[(size_t)m * KDIM + n] = acc[i][j][v] + bias[n];
            }
        }
    }
}

// ---------------------------------------------------------------------------
// Attention: one block per (b,h). 14 waves, one 16-row strip each.
// scores = q @ k^T (+rpb), softmax in registers, P @ v via per-wave LDS transpose
// ---------------------------------------------------------------------------
__global__ __launch_bounds__(448) void attn_kernel(
    const _Float16* __restrict__ q, const _Float16* __restrict__ k,
    const _Float16* __restrict__ vt, const float* __restrict__ rpb,
    _Float16* __restrict__ out)
{
    __shared__ _Float16 ks[NPAD * KSP];        // K tile   (32.3 KB)
    __shared__ _Float16 ps[14 * 16 * LDP];     // P stage  (17.5 KB)
    const int bh = blockIdx.x, h = bh % NH, b = bh / NH;
    const int t = threadIdx.x, lane = t & 31, w = t >> 5;
    const int half = lane >> 4, lr = lane & 15;
    const _Float16* kg = k + (size_t)bh * (NPAD * HD);
    const _Float16* vg = vt + (size_t)bh * (HD * NPAD);

    // stage K [224][64] -> LDS (pitch 72)
    #pragma unroll
    for (int i = 0; i < 4; ++i) {
        int c = t + i * 448;                   // 1792 chunks of 8 f16
        int r = c >> 3, cc = (c & 7) * 8;
        *(v8h*)(ks + r * KSP + cc) = *(const v8h*)(kg + r * HD + cc);
    }
    __syncthreads();

    const int m0 = w * 16;
    v8f sacc[14] = {};
    const _Float16* qg = q + ((size_t)bh * NPAD + m0) * HD;

    // scores = q @ k^T  (q A-frags straight from global; k B-frags from LDS)
    #pragma unroll
    for (int s = 0; s < 2; ++s) {
        const _Float16* ap = qg + lr * HD + s * 32 + 8 * half;
        v8h a1 = *(const v8h*)ap, a2 = *(const v8h*)(ap + 16);
        v16h af = MAKE_V16(a1, a2);
        #pragma unroll
        for (int j = 0; j < 14; ++j) {
            const _Float16* bp = ks + (j * 16 + lr) * KSP + s * 32 + 8 * half;
            v8h b1 = *(const v8h*)bp, b2 = *(const v8h*)(bp + 16);
            v16h bf = MAKE_V16(b1, b2);
            sacc[j] = __builtin_amdgcn_wmma_f32_16x16x32_f16(
                false, af, false, bf, (short)0, sacc[j], false, false);
        }
    }

    // + rel-pos bias, then row softmax (rows live on 16 lanes of one half-wave)
    const float* rb = rpb + (size_t)h * (NPAD * NPAD);
    #pragma unroll
    for (int v = 0; v < 8; ++v) {
        const int row = m0 + v + 8 * half;
        float mx = -3.0e38f;
        #pragma unroll
        for (int j = 0; j < 14; ++j) {
            float val = sacc[j][v] + rb[row * NPAD + j * 16 + lr];
            sacc[j][v] = val;
            mx = fmaxf(mx, val);
        }
        mx = fmaxf(mx, __shfl_xor(mx, 1, 32));
        mx = fmaxf(mx, __shfl_xor(mx, 2, 32));
        mx = fmaxf(mx, __shfl_xor(mx, 4, 32));
        mx = fmaxf(mx, __shfl_xor(mx, 8, 32));
        float sum = 0.f;
        #pragma unroll
        for (int j = 0; j < 14; ++j) {
            float e = __expf(sacc[j][v] - mx);
            sacc[j][v] = e;
            sum += e;
        }
        sum += __shfl_xor(sum, 1, 32);
        sum += __shfl_xor(sum, 2, 32);
        sum += __shfl_xor(sum, 4, 32);
        sum += __shfl_xor(sum, 8, 32);
        float inv = 1.0f / sum;
        #pragma unroll
        for (int j = 0; j < 14; ++j) sacc[j][v] *= inv;
    }

    // out = P @ v : transpose P through per-wave LDS strip, vT B-frags from global
    _Float16* pw = ps + w * (16 * LDP);
    v8f oacc[4] = {};
    #pragma unroll
    for (int s = 0; s < 7; ++s) {
        #pragma unroll
        for (int jj = 0; jj < 2; ++jj) {
            int j = 2 * s + jj;
            #pragma unroll
            for (int v = 0; v < 8; ++v)
                pw[(v + 8 * half) * LDP + jj * 16 + lr] = (_Float16)sacc[j][v];
        }
        asm volatile("s_wait_dscnt 0" ::: "memory");   // same-wave DS store->load
        const _Float16* ap = pw + lr * LDP + 8 * half;
        v8h a1 = *(const v8h*)ap, a2 = *(const v8h*)(ap + 16);
        v16h af = MAKE_V16(a1, a2);
        #pragma unroll
        for (int j2 = 0; j2 < 4; ++j2) {
            const _Float16* bp = vg + (j2 * 16 + lr) * NPAD + s * 32 + 8 * half;
            v8h b1 = *(const v8h*)bp, b2 = *(const v8h*)(bp + 16);
            v16h bf = MAKE_V16(b1, b2);
            oacc[j2] = __builtin_amdgcn_wmma_f32_16x16x32_f16(
                false, af, false, bf, (short)0, oacc[j2], false, false);
        }
    }
    #pragma unroll
    for (int j2 = 0; j2 < 4; ++j2)
        #pragma unroll
        for (int v = 0; v < 8; ++v) {
            int m = m0 + v + 8 * half;
            if (m < NT)
                out[((size_t)(b * NT + m)) * KDIM + h * HD + j2 * 16 + lr] = (_Float16)oacc[j2][v];
        }
}

// ---------------------------------------------------------------------------
// host launcher
// ---------------------------------------------------------------------------
extern "C" void kernel_launch(void* const* d_in, const int* in_sizes, int n_in,
                              void* d_out, int out_size, void* d_ws, size_t ws_size,
                              hipStream_t stream) {
    const float* x      = (const float*)d_in[0];
    const float* qkv_w  = (const float*)d_in[1];
    const float* q_bias = (const float*)d_in[2];
    const float* v_bias = (const float*)d_in[3];
    const float* proj_w = (const float*)d_in[4];
    const float* proj_b = (const float*)d_in[5];
    const float* rpb_h  = (const float*)d_in[6];
    const float* rpb_w  = (const float*)d_in[7];
    const int*   h_idx  = (const int*)d_in[8];
    const int*   w_idx  = (const int*)d_in[9];
    float* out = (float*)d_out;

    char* ws = (char*)d_ws;
    size_t off = 0;
    _Float16* xh   = (_Float16*)(ws + off); off += (size_t)MPAD * KDIM * 2;       // 19.46 MB
    _Float16* wqk  = (_Float16*)(ws + off); off += (size_t)NQKV * KDIM * 2;       //  3.54 MB
    _Float16* wpr  = (_Float16*)(ws + off); off += (size_t)KDIM * KDIM * 2;       //  1.18 MB
    const size_t perHead = (size_t)64 * NH * NPAD * HD;                            // 11,010,048 elems
    _Float16* qbuf = (_Float16*)(ws + off); off += perHead * 2;
    _Float16* kbuf = (_Float16*)(ws + off); off += perHead * 2;
    _Float16* vtbf = (_Float16*)(ws + off); off += perHead * 2;
    _Float16* abuf = (_Float16*)(ws + off); off += (size_t)MPAD * KDIM * 2;       // attn out (padded)
    float*    rpbb = (float*)(ws + off);    off += (size_t)NH * NPAD * NPAD * 4;

    // 0) zero q/k/vT region (pads must be benign)
    {
        int n4 = (int)(3 * perHead / 8);   // f16 count / 8 = float4 count
        zero_f4_kernel<<<(n4 + 255) / 256, 256, 0, stream>>>((float4*)qbuf, n4);
    }
    // 1) f32 -> f16 conversions
    {
        int tot = MPAD * KDIM;
        cvt_x_pad_kernel<<<(tot + 255) / 256, 256, 0, stream>>>(x, xh, MTOT, KDIM, MPAD);
        int nw = NQKV * KDIM;
        cvt_f16_kernel<<<(nw + 255) / 256, 256, 0, stream>>>(qkv_w, wqk, nw);
        int np = KDIM * KDIM;
        cvt_f16_kernel<<<(np + 255) / 256, 256, 0, stream>>>(proj_w, wpr, np);
    }
    // 2) combined rel-pos bias [H][224][224] with -inf key padding
    {
        int tot = NH * NPAD * NPAD;
        build_rpb_kernel<<<(tot + 255) / 256, 256, 0, stream>>>(rpb_h, rpb_w, h_idx, w_idx, rpbb);
    }
    // 3) fused QKV GEMM + head scatter (q scaled, biases applied)
    gemm_qkv_kernel<<<dim3(MPAD / 128, NQKV / 128), 256, 0, stream>>>(
        xh, wqk, q_bias, v_bias, qbuf, kbuf, vtbf);
    // 4) attention (scores + softmax + PV) per (b,h)
    attn_kernel<<<64 * NH, 448, 0, stream>>>(qbuf, kbuf, vtbf, rpbb, abuf);
    // 5) output projection
    gemm_proj_kernel<<<dim3(MPAD / 128, KDIM / 128), 256, 0, stream>>>(abuf, wpr, proj_b, out);
}
